// Glom_38757784879583
// MI455X (gfx1250) — compile-verified, matrix operand order checked by reading, not run on hardware
//
#include <hip/hip_runtime.h>
#include <hip/hip_bf16.h>
#include <math.h>
#include <stdint.h>

typedef __bf16 bf16;
typedef __attribute__((ext_vector_type(16))) __bf16 v16bf;
typedef __attribute__((ext_vector_type(8)))  __bf16 v8bf;
typedef __attribute__((ext_vector_type(8)))  float  v8f;

#define WMMA_BF16(a,b,c) \
  __builtin_amdgcn_wmma_f32_16x16x32_bf16(false,(a),false,(b),(short)0,(c),false,false)

// ---- problem constants --------------------------------------------------
// b=16, N=256 -> NTOK=4096 rows; D=512, H=2048, LEVELS=6, K padded 588->608
#define NTOK   4096
#define DIMD   512
#define HID    2048
#define KPAD   608
#define KREAL  588
#define NMAT   96          // b * LEVELS
#define SELF_VAL (-0.0005f)
#define TILE_BYTES 8192    // 128 rows x 32 bf16 = 128 x 64B

// ---- WMMA fragment load (from LDS tile or global) -----------------------
// 16-bit A-fragment layout (ISA 7.12.2): lane l (m = l&15, hi = l>>4),
// element e: K = (e&7) + 16*(e>>3) + 8*hi  -> two 16B chunks per lane.
__device__ __forceinline__ v16bf load_frag(const bf16* base, int ld, int row0,
                                           int k0, int lane) {
  int hi = (lane >> 4) & 1;
  int r  = row0 + (lane & 15);
  const bf16* p = base + (size_t)r * (size_t)ld + (size_t)(k0 + 8 * hi);
  union { v16bf v; v8bf h[2]; } u;
  u.h[0] = *(const v8bf*)(p);
  u.h[1] = *(const v8bf*)(p + 16);
  return u.v;
}

struct Acc { v8f c[2][4]; };

__device__ __forceinline__ void acc_zero(Acc& a) {
  v8f z = {0.f,0.f,0.f,0.f,0.f,0.f,0.f,0.f};
#pragma unroll
  for (int i = 0; i < 2; ++i)
#pragma unroll
    for (int j = 0; j < 4; ++j) a.c[i][j] = z;
}

// ---- async global -> LDS staging (CDNA5 ASYNCcnt path) ------------------
// One 128x32 bf16 tile = 512 x 16B chunks; 256 threads issue 2 chunks each.
__device__ __forceinline__ void issue_tile(const bf16* G, int ld, int row0,
                                           int k0, uint32_t lds_base) {
  int t = threadIdx.x;
#pragma unroll
  for (int u = 0; u < 2; ++u) {
    int ch  = t * 2 + u;
    int row = ch >> 2;          // 0..127
    int q   = ch & 3;           // 16B chunk within 64B row
    const bf16* gp = G + (size_t)(row0 + row) * (size_t)ld + (k0 + 8 * q);
    uint32_t lo = lds_base + (uint32_t)(row * 64 + q * 16);
    asm volatile("global_load_async_to_lds_b128 %0, %1, off"
                 :: "v"(lo), "v"(gp) : "memory");
  }
}

__device__ __forceinline__ void wait_async0() {
  asm volatile("s_wait_asynccnt 0" ::: "memory");
}

// Block = 256 threads = 8 waves (4 in M x 2 in N), 128x128 block tile.
// Double-buffered LDS staging: while WMMAs consume buffer `cur` (ds_load),
// the next K-slice is already in flight via async loads.
__device__ __forceinline__ void block_gemm(const bf16* A, int lda,
                                           const bf16* B, int ldb, int K,
                                           int bm, int bn,
                                           bf16* sA, bf16* sB, Acc& acc) {
  int lane = threadIdx.x & 31;
  int wid  = threadIdx.x >> 5;
  int wm   = (wid >> 1) * 32;
  int wn   = (wid & 1) * 64;
  uint32_t aBase = (uint32_t)(uintptr_t)sA;   // low 32 bits = LDS offset
  uint32_t bBase = (uint32_t)(uintptr_t)sB;
  int nk = K / 32;
  issue_tile(A, lda, bm, 0, aBase);
  issue_tile(B, ldb, bn, 0, bBase);
  for (int kk = 0; kk < nk; ++kk) {
    int cur = kk & 1;
    wait_async0();            // my chunks of buf[cur] have landed
    __syncthreads();          // everyone's chunks have landed / prev reads done
    if (kk + 1 < nk) {        // prefetch next slice into the other buffer
      issue_tile(A, lda, bm, (kk + 1) * 32, aBase + (cur ^ 1) * TILE_BYTES);
      issue_tile(B, ldb, bn, (kk + 1) * 32, bBase + (cur ^ 1) * TILE_BYTES);
    }
    const bf16* pA = sA + cur * (TILE_BYTES / 2);
    const bf16* pB = sB + cur * (TILE_BYTES / 2);
    v16bf a0 = load_frag(pA, 32, wm,      0, lane);
    v16bf a1 = load_frag(pA, 32, wm + 16, 0, lane);
    v16bf b0 = load_frag(pB, 32, wn,      0, lane);
    v16bf b1 = load_frag(pB, 32, wn + 16, 0, lane);
    v16bf b2 = load_frag(pB, 32, wn + 32, 0, lane);
    v16bf b3 = load_frag(pB, 32, wn + 48, 0, lane);
    acc.c[0][0] = WMMA_BF16(a0, b0, acc.c[0][0]);
    acc.c[0][1] = WMMA_BF16(a0, b1, acc.c[0][1]);
    acc.c[0][2] = WMMA_BF16(a0, b2, acc.c[0][2]);
    acc.c[0][3] = WMMA_BF16(a0, b3, acc.c[0][3]);
    acc.c[1][0] = WMMA_BF16(a1, b0, acc.c[1][0]);
    acc.c[1][1] = WMMA_BF16(a1, b1, acc.c[1][1]);
    acc.c[1][2] = WMMA_BF16(a1, b2, acc.c[1][2]);
    acc.c[1][3] = WMMA_BF16(a1, b3, acc.c[1][3]);
  }
}

#define GEMM_SHARED                                                          \
  __shared__ __align__(16) bf16 sA[2 * TILE_BYTES / 2];                      \
  __shared__ __align__(16) bf16 sB[2 * TILE_BYTES / 2];

#define GEMM_COORDS                                                          \
  int lane = threadIdx.x & 31;                                               \
  int wid  = threadIdx.x >> 5;                                               \
  int m0 = blockIdx.y * 128 + (wid >> 1) * 32;                               \
  int n0 = blockIdx.x * 128 + (wid & 1) * 64;

// C layout (ISA 7.12.2): lane l -> col = n0 + (l&15); rows m0 + 8*(l>>4) + v
#define EPILOGUE_LOOP(BODY)                                                  \
  _Pragma("unroll")                                                          \
  for (int i = 0; i < 2; ++i) {                                              \
    _Pragma("unroll")                                                        \
    for (int j = 0; j < 4; ++j) {                                            \
      int col   = n0 + 16 * j + (lane & 15);                                 \
      int rbase = m0 + 16 * i + 8 * (lane >> 4);                             \
      _Pragma("unroll")                                                      \
      for (int v = 0; v < 8; ++v) {                                          \
        int row = rbase + v;                                                 \
        float y = acc.c[i][j][v];                                            \
        BODY;                                                                \
      }                                                                      \
    }                                                                        \
  }

// ---- setup kernels ------------------------------------------------------

__global__ __launch_bounds__(256)
void k_init_levels(const float* __restrict__ initl, float* __restrict__ levels) {
  size_t idx = (size_t)blockIdx.x * 256 + threadIdx.x;
  if (idx >= (size_t)NTOK * 6 * DIMD) return;
  levels[idx] = initl[idx % (6 * DIMD)];
}

// generic f32 (K x N) -> bf16 transposed (N x K)
__global__ __launch_bounds__(256)
void k_transpose(const float* __restrict__ src, bf16* __restrict__ dst,
                 int K, int N) {
  size_t idx = (size_t)blockIdx.x * 256 + threadIdx.x;
  if (idx >= (size_t)K * N) return;
  int n = (int)(idx / K);
  int k = (int)(idx % K);
  dst[idx] = (bf16)src[(size_t)k * N + n];
}

// tok_w (588 x 512) -> bf16 (512 x 608) with zero K-padding
__global__ __launch_bounds__(256)
void k_transpose_tokw(const float* __restrict__ src, bf16* __restrict__ dst) {
  size_t idx = (size_t)blockIdx.x * 256 + threadIdx.x;
  if (idx >= (size_t)DIMD * KPAD) return;
  int n = (int)(idx / KPAD);
  int k = (int)(idx % KPAD);
  dst[idx] = (k < KREAL) ? (bf16)src[(size_t)k * DIMD + n] : (bf16)0.0f;
}

// img (16,3,224,224) -> patch matrix (4096 x 608) bf16, feature=(dy*14+dx)*3+c
__global__ __launch_bounds__(256)
void k_patches(const float* __restrict__ img, bf16* __restrict__ patches) {
  size_t idx = (size_t)blockIdx.x * 256 + threadIdx.x;
  if (idx >= (size_t)NTOK * KPAD) return;
  int f = (int)(idx % KPAD);
  int t = (int)(idx / KPAD);
  bf16 v = (bf16)0.0f;
  if (f < KREAL) {
    int dy = f / 42, rem = f % 42, dx = rem / 3, c = rem % 3;
    int b = t >> 8, n = t & 255, py = n >> 4, px = n & 15;
    v = (bf16)img[(((size_t)b * 3 + c) * 224 + (py * 14 + dy)) * 224 +
                  (px * 14 + dx)];
  }
  patches[idx] = v;
}

// ---- GEMM kernels -------------------------------------------------------

__global__ __launch_bounds__(256)
void k_gemm_tokens(const bf16* __restrict__ P, const bf16* __restrict__ Wt,
                   const float* __restrict__ bias, bf16* __restrict__ X0) {
  GEMM_SHARED; GEMM_COORDS;
  Acc acc; acc_zero(acc);
  block_gemm(P, KPAD, Wt, KPAD, KPAD, blockIdx.y * 128, blockIdx.x * 128,
             sA, sB, acc);
  EPILOGUE_LOOP({
    X0[(size_t)row * DIMD + col] = (bf16)(y + bias[col]);
  })
}

__global__ __launch_bounds__(256)
void k_ff1(const bf16* __restrict__ Xg, const bf16* __restrict__ W1Tg,
           const float* __restrict__ b1g, bf16* __restrict__ H) {
  GEMM_SHARED; GEMM_COORDS;
  Acc acc; acc_zero(acc);
  block_gemm(Xg, DIMD, W1Tg, DIMD, DIMD, blockIdx.y * 128, blockIdx.x * 128,
             sA, sB, acc);
  EPILOGUE_LOOP({
    float t = y + b1g[col];
    t = 0.5f * t * (1.0f + erff(t * 0.70710678118654752f));  // exact GELU
    H[(size_t)row * HID + col] = (bf16)t;
  })
}

__global__ __launch_bounds__(256)
void k_ff2(const bf16* __restrict__ H, const bf16* __restrict__ W2Tg,
           const float* __restrict__ b2g, float* __restrict__ accb, int lvl) {
  GEMM_SHARED; GEMM_COORDS;
  Acc acc; acc_zero(acc);
  block_gemm(H, HID, W2Tg, HID, HID, blockIdx.y * 128, blockIdx.x * 128,
             sA, sB, acc);
  EPILOGUE_LOOP({
    size_t o = ((size_t)row * 6 + lvl) * DIMD + col;
    accb[o] += y + b2g[col];
  })
}

// sim = L @ K^T * d^-0.5   (per (b,level) matrix, z = blockIdx.z)
__global__ __launch_bounds__(256)
void k_gemm_sim(const bf16* __restrict__ Lbf, const bf16* __restrict__ Kbf,
                float* __restrict__ simb) {
  GEMM_SHARED; GEMM_COORDS;
  int z = blockIdx.z;
  const bf16* A = Lbf + (size_t)z * 256 * DIMD;
  const bf16* B = Kbf + (size_t)z * 256 * DIMD;  // row n of Kbf == col n of K^T
  Acc acc; acc_zero(acc);
  block_gemm(A, DIMD, B, DIMD, DIMD, blockIdx.y * 128, blockIdx.x * 128,
             sA, sB, acc);
  EPILOGUE_LOOP({
    simb[(size_t)z * 65536 + (size_t)row * 256 + col] =
        y * 0.044194173824159216f;  // 512^-0.5
  })
}

// cons: acc = levels + attn @ L   (initializes the per-iteration accumulator)
__global__ __launch_bounds__(256)
void k_gemm_cons(const bf16* __restrict__ attnb, const bf16* __restrict__ LbfT,
                 const float* __restrict__ levels, float* __restrict__ accb) {
  GEMM_SHARED; GEMM_COORDS;
  int z = blockIdx.z, bb = z / 6, l = z % 6;
  const bf16* A = attnb + (size_t)z * 65536;        // 256 x 256
  const bf16* B = LbfT + (size_t)z * DIMD * 256;    // D x 256 (N x K)
  Acc acc; acc_zero(acc);
  block_gemm(A, 256, B, 256, 256, blockIdx.y * 128, blockIdx.x * 128,
             sA, sB, acc);
  EPILOGUE_LOOP({
    size_t o = (((size_t)bb * 256 + row) * 6 + l) * DIMD + col;
    accb[o] = levels[o] + y;
  })
}

// ---- per-iteration elementwise / reduction kernels ----------------------

// groups 1..5: bf16(levels[l-1]); groups 6..10: bf16(levels[g-5] + pos)
__global__ __launch_bounds__(256)
void k_prep_X(const float* __restrict__ levels, const float* __restrict__ pos,
              bf16* __restrict__ X) {
  size_t idx = (size_t)blockIdx.x * 256 + threadIdx.x;
  if (idx >= (size_t)10 * NTOK * DIMD) return;
  int d = (int)(idx & (DIMD - 1));
  size_t r = idx >> 9;
  int t  = (int)(r % NTOK);
  int g  = (int)(r / NTOK) + 1;
  float v;
  if (g <= 5) v = levels[((size_t)t * 6 + (g - 1)) * DIMD + d];
  else        v = levels[((size_t)t * 6 + (g - 5)) * DIMD + d] +
                  pos[(size_t)(t & 255) * DIMD + d];
  X[((size_t)g * NTOK + t) * DIMD + d] = (bf16)v;
}

// Per (b,level,token): write L (bf16), K = L*rsqrt(max(sum L^2,1e-12)) (bf16),
// and L^T (bf16, D x 256) for the cons GEMM's B operand.
__global__ __launch_bounds__(256)
void k_cons_prep(const float* __restrict__ levels, bf16* __restrict__ Lbf,
                 bf16* __restrict__ Kbf, bf16* __restrict__ LbfT) {
  __shared__ float red[256];
  int m = blockIdx.y, i = blockIdx.x, t = threadIdx.x;
  int bb = m / 6, l = m % 6;
  size_t base = (((size_t)bb * 256 + i) * 6 + l) * DIMD;
  float x0 = levels[base + t];
  float x1 = levels[base + t + 256];
  red[t] = x0 * x0 + x1 * x1;
  __syncthreads();
  for (int s = 128; s > 0; s >>= 1) {
    if (t < s) red[t] += red[t + s];
    __syncthreads();
  }
  float rs = rsqrtf(fmaxf(red[0], 1e-12f));
  size_t lb = ((size_t)m * 256 + i) * DIMD;
  Lbf[lb + t] = (bf16)x0;              Lbf[lb + t + 256] = (bf16)x1;
  Kbf[lb + t] = (bf16)(x0 * rs);       Kbf[lb + t + 256] = (bf16)(x1 * rs);
  size_t tb = (size_t)m * DIMD * 256;
  LbfT[tb + (size_t)t * 256 + i]         = (bf16)x0;
  LbfT[tb + (size_t)(t + 256) * 256 + i] = (bf16)x1;
}

__global__ __launch_bounds__(256)
void k_softmax(const float* __restrict__ simb, bf16* __restrict__ attnb) {
  __shared__ float red[256];
  int m = blockIdx.y, i = blockIdx.x, j = threadIdx.x;
  size_t o = ((size_t)m * 256 + i) * 256;
  float v = simb[o + j];
  if (j == i) v = SELF_VAL;            // diagonal replaced pre-softmax
  red[j] = v; __syncthreads();
  for (int s = 128; s > 0; s >>= 1) {
    if (j < s) red[j] = fmaxf(red[j], red[j + s]);
    __syncthreads();
  }
  float mx = red[0]; __syncthreads();
  float e = expf(v - mx);
  red[j] = e; __syncthreads();
  for (int s = 128; s > 0; s >>= 1) {
    if (j < s) red[j] += red[j + s];
    __syncthreads();
  }
  attnb[o + j] = (bf16)(e / red[0]);
}

// levels = acc * inv_contrib  (1/4 for levels 0..4, 1/3 for level 5)
__global__ __launch_bounds__(256)
void k_update(const float* __restrict__ accb, float* __restrict__ levels) {
  size_t idx = (size_t)blockIdx.x * 256 + threadIdx.x;
  if (idx >= (size_t)NTOK * 6 * DIMD) return;
  int l = (int)((idx >> 9) % 6);
  levels[idx] = accb[idx] * (l == 5 ? (1.0f / 3.0f) : 0.25f);
}

// ---- host driver --------------------------------------------------------

extern "C" void kernel_launch(void* const* d_in, const int* in_sizes, int n_in,
                              void* d_out, int out_size, void* d_ws,
                              size_t ws_size, hipStream_t stream) {
  (void)in_sizes; (void)n_in; (void)out_size; (void)ws_size;
  const float* img   = (const float*)d_in[0];
  const float* tok_w = (const float*)d_in[1];
  const float* tok_b = (const float*)d_in[2];
  const float* pos   = (const float*)d_in[3];
  const float* initl = (const float*)d_in[4];
  const float* bu_w1 = (const float*)d_in[5];
  const float* bu_b1 = (const float*)d_in[6];
  const float* bu_w2 = (const float*)d_in[7];
  const float* bu_b2 = (const float*)d_in[8];
  const float* td_w1 = (const float*)d_in[9];
  const float* td_b1 = (const float*)d_in[10];
  const float* td_w2 = (const float*)d_in[11];
  const float* td_b2 = (const float*)d_in[12];
  float* levels = (float*)d_out;  // levels live in d_out throughout

  char* ws = (char*)d_ws;
  size_t off = 0;
  auto alloc = [&](size_t bytes) -> char* {
    char* p = ws + off;
    off += (bytes + 255) & ~(size_t)255;
    return p;
  };
  bf16*  tokwT   = (bf16*)alloc((size_t)DIMD * KPAD * 2);
  bf16*  patches = (bf16*)alloc((size_t)NTOK * KPAD * 2);
  bf16*  X       = (bf16*)alloc((size_t)11 * NTOK * DIMD * 2);
  bf16*  W1T     = (bf16*)alloc((size_t)11 * HID * DIMD * 2);
  bf16*  W2T     = (bf16*)alloc((size_t)11 * DIMD * HID * 2);
  bf16*  H       = (bf16*)alloc((size_t)NTOK * HID * 2);
  float* accb    = (float*)alloc((size_t)NTOK * 6 * DIMD * 4);
  bf16*  Lbf     = (bf16*)alloc((size_t)NMAT * 256 * DIMD * 2);
  bf16*  Kbf     = (bf16*)alloc((size_t)NMAT * 256 * DIMD * 2);
  bf16*  LbfT    = (bf16*)alloc((size_t)NMAT * DIMD * 256 * 2);
  float* simb    = (float*)alloc((size_t)NMAT * 256 * 256 * 4);
  bf16*  attnb   = (bf16*)alloc((size_t)NMAT * 256 * 256 * 2);

  dim3 blk(256);

  k_init_levels<<<dim3((NTOK * 6 * DIMD) / 256), blk, 0, stream>>>(initl, levels);
  k_transpose_tokw<<<dim3((DIMD * KPAD + 255) / 256), blk, 0, stream>>>(tok_w, tokwT);
  k_patches<<<dim3((NTOK * KPAD + 255) / 256), blk, 0, stream>>>(img, patches);
  for (int g = 0; g < 11; ++g) {
    const float* s1 = g < 6 ? bu_w1 + (size_t)g * DIMD * HID
                            : td_w1 + (size_t)(g - 6) * DIMD * HID;
    const float* s2 = g < 6 ? bu_w2 + (size_t)g * HID * DIMD
                            : td_w2 + (size_t)(g - 6) * HID * DIMD;
    k_transpose<<<dim3((DIMD * HID) / 256), blk, 0, stream>>>(
        s1, W1T + (size_t)g * HID * DIMD, DIMD, HID);
    k_transpose<<<dim3((HID * DIMD) / 256), blk, 0, stream>>>(
        s2, W2T + (size_t)g * DIMD * HID, HID, DIMD);
  }
  // tokens -> X group 0 (constant "bottom" input for all iterations)
  k_gemm_tokens<<<dim3(DIMD / 128, NTOK / 128), blk, 0, stream>>>(
      patches, tokwT, tok_b, X);

  for (int it = 0; it < 12; ++it) {
    k_prep_X<<<dim3((10 * NTOK * DIMD) / 256), blk, 0, stream>>>(levels, pos, X);
    k_cons_prep<<<dim3(256, NMAT), blk, 0, stream>>>(levels, Lbf, Kbf, LbfT);
    k_gemm_sim<<<dim3(2, 2, NMAT), blk, 0, stream>>>(Lbf, Kbf, simb);
    k_softmax<<<dim3(256, NMAT), blk, 0, stream>>>(simb, attnb);
    k_gemm_cons<<<dim3(DIMD / 128, 2, NMAT), blk, 0, stream>>>(attnb, LbfT,
                                                               levels, accb);
    for (int g = 0; g < 11; ++g) {
      const float* b1g = g < 6 ? bu_b1 + (size_t)g * HID
                               : td_b1 + (size_t)(g - 6) * HID;
      const float* b2g = g < 6 ? bu_b2 + (size_t)g * DIMD
                               : td_b2 + (size_t)(g - 6) * DIMD;
      int lvl = g < 6 ? g : g - 6;
      k_ff1<<<dim3(HID / 128, NTOK / 128), blk, 0, stream>>>(
          X + (size_t)g * NTOK * DIMD, W1T + (size_t)g * HID * DIMD, b1g, H);
      k_ff2<<<dim3(DIMD / 128, NTOK / 128), blk, 0, stream>>>(
          H, W2T + (size_t)g * DIMD * HID, b2g, accb, lvl);
    }
    k_update<<<dim3((NTOK * 6 * DIMD) / 256), blk, 0, stream>>>(accb, levels);
  }
}
